// Blend_83296595739329
// MI455X (gfx1250) — matrix operand order repaired and checked by reading, not compiled, not run on hardware
//
#include <hip/hip_runtime.h>
#include <hip/hip_bf16.h>

// ---------------------------------------------------------------------------
// MI455X (gfx1250) implementation, round 4.
// Compute-bound (~0.3 TFLOP/call vs ~15us of HBM at 23.3 TB/s) -> all convs as
// implicit GEMM on v_wmma_f32_16x16x32_f16 (wave32).
//  * Activations: f16 interleaved [pixel][channel] rows -> B frag = 2x b128,
//    D store = 1x b128 per tile.
//  * Weights: packed f16 [Mpad][tap*Cin+c]; staged once per workgroup into
//    LDS by the Tensor Data Mover (tensor_load_to_lds + s_wait_tensorcnt),
//    A fragments then come from ds_load_b128 (zero global weight traffic).
//  * Dense-block concats = column windows (stride, offset) of shared regions.
// ---------------------------------------------------------------------------

#define HH 256
#define WW 256
#define BB 4
#define HWSZ (HH * WW)           // 65536
#define NP   (BB * HWSZ)         // 262144 pixels (batch folded in)

typedef __attribute__((ext_vector_type(16))) _Float16 v16h;
typedef __attribute__((ext_vector_type(8)))  _Float16 h8;
typedef __attribute__((ext_vector_type(8)))  float    v8f;
typedef __attribute__((ext_vector_type(4)))  unsigned int u32x4;
typedef __attribute__((ext_vector_type(8)))  int i32x8;
typedef __attribute__((ext_vector_type(4)))  int i32x4;

#define SHUF16(a, b) __builtin_shufflevector(a, b, 0, 1, 2, 3, 4, 5, 6, 7, 8, 9, 10, 11, 12, 13, 14, 15)
#define SMEMB(MT, KT) ((MT) * 16 * (KT) * 2)   // LDS bytes for packed weights

// ---------------------------------------------------------------------------
// Weight packing: f32 OIHW -> f16 [Mpad][tap*Cin + c], zero-padded rows.
// ---------------------------------------------------------------------------
__global__ void pack_w_kernel(const float* __restrict__ src, _Float16* __restrict__ dst,
                              int Cout, int Cin, int KS, int Mpad) {
  int Ktot = KS * KS * Cin;
  int idx = blockIdx.x * blockDim.x + threadIdx.x;
  if (idx >= Mpad * Ktot) return;
  int m = idx / Ktot;
  int k = idx - m * Ktot;
  int tap = k / Cin;
  int c = k - tap * Cin;
  int ky = tap / KS, kx = tap % KS;
  float v = 0.f;
  if (m < Cout) v = src[((m * Cin + c) * KS + ky) * KS + kx];
  dst[idx] = (_Float16)v;
}

// Pr[v] (3,128,128) -> f16 [128][128] 1x1 pack; v read from device scalar.
__global__ void pack_pr_kernel(const float* __restrict__ pr, const int* __restrict__ vp,
                               _Float16* __restrict__ dst) {
  int v = *vp;
  int idx = blockIdx.x * blockDim.x + threadIdx.x;
  if (idx >= 128 * 128) return;
  dst[idx] = (_Float16)pr[v * 128 * 128 + idx];
}

// Strided column-window copy: dst[p][dof..dof+ncols) = src[p][so..so+ncols)
__global__ void copy_cols_kernel(const _Float16* __restrict__ src, int ss, int so,
                                 _Float16* __restrict__ dst, int ds, int dof, int ncols) {
  int p = blockIdx.x * 256 + threadIdx.x;
  const _Float16* s = src + (size_t)p * ss + so;
  _Float16* d = dst + (size_t)p * ds + dof;
  for (int c = 0; c < ncols; c += 8) *(h8*)(d + c) = *(const h8*)(s + c);
}

// ---------------------------------------------------------------------------
// Feature extraction conv1: 1 -> 32, 3x3, pad 1, leaky relu (K=9, VALU).
// dst interleaved [p][32].
// ---------------------------------------------------------------------------
__global__ void fe1_kernel(const float* __restrict__ x, const float* __restrict__ w,
                           const float* __restrict__ bias, _Float16* __restrict__ dst) {
  int p = blockIdx.x * 256 + threadIdx.x;
  int b = p >> 16;
  int rem = p & 65535;
  int y = rem >> 8;
  int xx = rem & 255;
  float in[9];
#pragma unroll
  for (int ky = 0; ky < 3; ++ky)
#pragma unroll
    for (int kx = 0; kx < 3; ++kx) {
      int yy = y + ky - 1, xc = xx + kx - 1;
      bool ok = (yy >= 0) && (yy < HH) && (xc >= 0) && (xc < WW);
      in[ky * 3 + kx] = ok ? x[b * HWSZ + yy * WW + xc] : 0.f;
    }
  h8 outv[4];
#pragma unroll
  for (int co = 0; co < 32; ++co) {
    float s = bias[co];
#pragma unroll
    for (int t = 0; t < 9; ++t) s += w[co * 9 + t] * in[t];
    s = s > 0.f ? s : 0.2f * s;
    outv[co >> 3][co & 7] = (_Float16)s;
  }
  _Float16* d = dst + (size_t)p * 32;
#pragma unroll
  for (int q = 0; q < 4; ++q) *(h8*)(d + q * 8) = outv[q];
}

// ---------------------------------------------------------------------------
// WMMA implicit-GEMM convolution; weights staged in LDS via TDM.
//   src : f16 rows [p][sstride], channels [soff, soff+CIN)
//   wpk : f16 [MTILES*16][KS*KS*CIN]  (tap-major K)
//   dst : f16 rows [p][dstride], channels [doff, doff+MTILES*16)
// Fragment layouts used here (wave32):
//   A (16x32 f16): row = lane&15; halves 0..7 -> K = 16*(lane>=16)+0..7,
//                  halves 8..15 -> K = 16*(lane>=16)+8..15  -> 2x ds b128.
//   B (32x16 f16): col = lane&15; half h -> K = 16*(lane>=16)+h -> 2x b128.
//   C/D (16x16 f32): row r -> M = r + 8*(lane>=16); col = lane&15 -> 1x b128.
// Grids are exact multiples of the block tile, so EXEC is all-ones for WMMA.
// ---------------------------------------------------------------------------
template <int CIN, int MTILES, int NTILES, int KS, bool RELU>
__global__ __launch_bounds__(256) void conv_wmma_kernel(
    const _Float16* __restrict__ src, int sstride, int soff,
    const _Float16* __restrict__ wpk, const float* __restrict__ bias,
    _Float16* __restrict__ dst, int dstride, int doff, int Cout) {
  constexpr int KTOT = KS * KS * CIN;
  constexpr int WELEMS = MTILES * 16 * KTOT;
  extern __shared__ _Float16 ldsw[];  // WELEMS packed weights

  const int lane = threadIdx.x & 31;
  const int wave = threadIdx.x >> 5;
  const int col = lane & 15;
  const int hi = lane >> 4;
  const int hiB = hi << 4;
  const int hiA = hi << 4;

  // ---- stage packed weights into LDS with the Tensor Data Mover (wave 0)
#if __has_builtin(__builtin_amdgcn_tensor_load_to_lds)
  if (wave == 0) {
    unsigned lds_addr = (unsigned)(size_t)(&ldsw[0]);  // LDS aperture: low 32b
    unsigned long long ga = (unsigned long long)(size_t)wpk;
    unsigned n8 = (unsigned)(WELEMS * 2 / 8);  // payload in 8-byte units
    // D# group0: count=1 | lds_addr | global_addr[56:0] | type=2
    u32x4 g0 = {1u, lds_addr, (unsigned)ga,
                (unsigned)((ga >> 32) & 0x1ffffffu) | (2u << 30)};
    // D# group1: data_size=3 (8B); 1-D tile: tensor_dim0=tile_dim0=n8,
    // tensor_dim1=tile_dim1=1, tile_dim2=0, dim0_stride=n8, dim1_stride=0.
    i32x8 g1 = {(int)(3u << 16),                       // mask=0, data_size=3
                (int)((n8 & 0xffffu) << 16),           // tensor_dim0[15:0]
                (int)(((n8 >> 16) & 0xffffu) | (1u << 16)),  // td0 hi | td1 lo
                (int)((n8 & 0xffffu) << 16),           // td1 hi=0 | tile_dim0
                1,                                     // tile_dim1=1, tile_dim2=0
                (int)n8,                               // dim0_stride[31:0]
                0, 0};                                 // stride hi, dim1_stride
    i32x4 gz = {0, 0, 0, 0};                           // groups 2/3 unused (<=2D)
    i32x8 gz8 = {0, 0, 0, 0, 0, 0, 0, 0};              // extra group (6-arg form)
    __builtin_amdgcn_tensor_load_to_lds(g0, g1, gz, gz, gz8, 0);
    __builtin_amdgcn_s_wait_tensorcnt(0);
  }
#else
  // fallback: cooperative copy (keeps file portable across toolchains)
  for (int i = threadIdx.x * 8; i < WELEMS; i += 256 * 8)
    *(h8*)(ldsw + i) = *(const h8*)(wpk + i);
#endif
  __syncthreads();

  const int p0 = (blockIdx.x * 8 + wave) * (16 * NTILES);
  int pixn[NTILES], pyn[NTILES], pxn[NTILES], rowb[NTILES];
#pragma unroll
  for (int n = 0; n < NTILES; ++n) {
    int p = p0 + n * 16 + col;
    pixn[n] = p;
    int rem = p & 65535;
    pyn[n] = rem >> 8;
    pxn[n] = rem & 255;
    rowb[n] = (p >> 16) * HWSZ;
  }

  const v16h z16 = {0, 0, 0, 0, 0, 0, 0, 0, 0, 0, 0, 0, 0, 0, 0, 0};
  const v8f zf = {0.f, 0.f, 0.f, 0.f, 0.f, 0.f, 0.f, 0.f};
  v8f acc[MTILES][NTILES];
#pragma unroll
  for (int t = 0; t < MTILES; ++t)
#pragma unroll
    for (int n = 0; n < NTILES; ++n) acc[t][n] = zf;

  for (int k0 = 0; k0 < KTOT; k0 += 32) {
    const int tap = (KS == 1) ? 0 : (k0 / CIN);
    const int c0 = (KS == 1) ? k0 : (k0 - tap * CIN);
    const int dy = tap / KS - KS / 2;
    const int dx = tap % KS - KS / 2;

    // ---- B fragments: 2x b128 contiguous channel loads per pixel tile
    v16h bf[NTILES];
#pragma unroll
    for (int n = 0; n < NTILES; ++n) {
      const int yy = pyn[n] + dy;
      const int xx = pxn[n] + dx;
      const bool ok = (yy >= 0) & (yy < HH) & (xx >= 0) & (xx < WW);
      const int sp = ok ? (rowb[n] + yy * WW + xx) : 0;
      const _Float16* br = src + (size_t)sp * sstride + soff + c0 + hiB;
      h8 b0 = *(const h8*)br;
      h8 b1 = *(const h8*)(br + 8);
      v16h bb = SHUF16(b0, b1);
      bf[n] = ok ? bb : z16;
    }

    // ---- A fragments from LDS (2x ds b128) + WMMA, A reused across NTILES
#pragma unroll
    for (int t = 0; t < MTILES; ++t) {
      const _Float16* wr = ldsw + (size_t)(t * 16 + col) * KTOT + k0 + hiA;
      h8 a0 = *(const h8*)wr;
      h8 a1 = *(const h8*)(wr + 8);
      v16h af = SHUF16(a0, a1);
#pragma unroll
      for (int n = 0; n < NTILES; ++n)
        acc[t][n] = __builtin_amdgcn_wmma_f32_16x16x32_f16(
            false, af, false, bf[n], (short)0, acc[t][n], false, false);
    }
  }

  // ---- epilogue: bias + leaky-relu, one b128 store per tile
#pragma unroll
  for (int t = 0; t < MTILES; ++t)
#pragma unroll
    for (int n = 0; n < NTILES; ++n) {
      h8 ov;
#pragma unroll
      for (int r = 0; r < 8; ++r) {
        const int m = t * 16 + (hi << 3) + r;
        float val = acc[t][n][r];
        if (bias != nullptr && m < Cout) val += bias[m];
        if (RELU) val = val > 0.f ? val : 0.2f * val;
        ov[r] = (_Float16)val;
      }
      *(h8*)(dst + (size_t)pixn[n] * dstride + doff + t * 16 + (hi << 3)) = ov;
    }
}

// ---------------------------------------------------------------------------
// Warp: bilinear-warp per-frame features (32 ch f16, interleaved rows) into
// the xc region columns [f*32, f*32+32), and inputs (1 ch f32, planar) into
// inp[f][p]. v-dependent pass-through per reference. blockIdx.y = frame.
// ---------------------------------------------------------------------------
__global__ void warp_kernel(const float* __restrict__ xall, const float* __restrict__ flows,
                            const int* __restrict__ vp, const _Float16* __restrict__ feats,
                            _Float16* __restrict__ xcb, float* __restrict__ inp) {
  int p = blockIdx.x * 256 + threadIdx.x;
  int f = blockIdx.y;
  int v = *vp;
  bool dw = (v == 1) ? (f <= 1) : (v == 2) ? (f != 1 && f != 3) : true;
  int b = p >> 16;
  int rem = p & 65535;
  int y = rem >> 8;
  int x = rem & 255;
  const _Float16* fsrc = feats + (size_t)f * NP * 32;  // [p][32]
  _Float16* drow = xcb + (size_t)p * 128 + f * 32;
  if (dw) {
    float gx = flows[(size_t)((f * BB + b) * 2 + 0) * HWSZ + rem] + (float)x;
    float gy = flows[(size_t)((f * BB + b) * 2 + 1) * HWSZ + rem] + (float)y;
    float fx0 = floorf(gx), fy0 = floorf(gy);
    float wx = gx - fx0, wy = gy - fy0;
    int x0 = (int)fx0, y0 = (int)fy0;
    int idxs[4];
    float wv[4];
    float wgt[4] = {(1.f - wx) * (1.f - wy), wx * (1.f - wy), (1.f - wx) * wy, wx * wy};
#pragma unroll
    for (int cy = 0; cy < 2; ++cy)
#pragma unroll
      for (int cx = 0; cx < 2; ++cx) {
        int yy = y0 + cy, xq = x0 + cx;
        bool ok = (xq >= 0) && (xq < WW) && (yy >= 0) && (yy < HH);
        int yc = yy < 0 ? 0 : (yy > HH - 1 ? HH - 1 : yy);
        int xc = xq < 0 ? 0 : (xq > WW - 1 ? WW - 1 : xq);
        idxs[cy * 2 + cx] = b * HWSZ + yc * WW + xc;
        wv[cy * 2 + cx] = ok ? wgt[cy * 2 + cx] : 0.f;
      }
    const _Float16* r0 = fsrc + (size_t)idxs[0] * 32;
    const _Float16* r1 = fsrc + (size_t)idxs[1] * 32;
    const _Float16* r2 = fsrc + (size_t)idxs[2] * 32;
    const _Float16* r3 = fsrc + (size_t)idxs[3] * 32;
    h8 outv[4];
#pragma unroll
    for (int c = 0; c < 32; ++c) {
      float s = (float)r0[c] * wv[0] + (float)r1[c] * wv[1] +
                (float)r2[c] * wv[2] + (float)r3[c] * wv[3];
      outv[c >> 3][c & 7] = (_Float16)s;
    }
#pragma unroll
    for (int q = 0; q < 4; ++q) *(h8*)(drow + q * 8) = outv[q];
    const float* xs = xall + (size_t)f * NP;
    inp[(size_t)f * NP + p] =
        xs[idxs[0]] * wv[0] + xs[idxs[1]] * wv[1] + xs[idxs[2]] * wv[2] + xs[idxs[3]] * wv[3];
  } else {
    const _Float16* srow = fsrc + (size_t)p * 32;
#pragma unroll
    for (int q = 0; q < 4; ++q) *(h8*)(drow + q * 8) = *(const h8*)(srow + q * 8);
    inp[(size_t)f * NP + p] = xall[(size_t)f * NP + p];
  }
}

// ---------------------------------------------------------------------------
// Dynamic filter: out[p] = sum_{c<4,tap<9} ex[p][c*9+tap] * inp_pad[c][y+dy][x+dx]
// ex interleaved [p][48] (36 valid), inp planar f32 [4][NP].
// ---------------------------------------------------------------------------
__global__ void dynfilter_kernel(const _Float16* __restrict__ ex, const float* __restrict__ inp,
                                 float* __restrict__ out) {
  int p = blockIdx.x * 256 + threadIdx.x;
  int b = p >> 16;
  int rem = p & 65535;
  int y = rem >> 8;
  int x = rem & 255;
  const _Float16* er = ex + (size_t)p * 48;
  float acc = 0.f;
#pragma unroll
  for (int c = 0; c < 4; ++c)
#pragma unroll
    for (int ky = 0; ky < 3; ++ky)
#pragma unroll
      for (int kx = 0; kx < 3; ++kx) {
        int yy = y + ky - 1, xx = x + kx - 1;
        bool ok = (yy >= 0) && (yy < HH) && (xx >= 0) && (xx < WW);
        float val = ok ? inp[(size_t)c * NP + b * HWSZ + yy * WW + xx] : 0.f;
        acc += val * (float)er[c * 9 + ky * 3 + kx];
      }
  out[p] = acc;
}

// ---------------------------------------------------------------------------
// Host-side orchestration.
// ---------------------------------------------------------------------------
extern "C" void kernel_launch(void* const* d_in, const int* in_sizes, int n_in,
                              void* d_out, int out_size, void* d_ws, size_t ws_size,
                              hipStream_t stream) {
  (void)in_sizes; (void)n_in; (void)out_size; (void)ws_size;
  const size_t PL = NP;

  // ---- inputs (setup_inputs dict order)
  const float* x_all = (const float*)d_in[0];   // (4,B,1,H,W)
  const float* flows = (const float*)d_in[1];   // (4,B,2,H,W)
  const int* vptr    = (const int*)d_in[2];     // scalar v
  const float* pr_w = (const float*)d_in[19];
  const float* c1_w = (const float*)d_in[20];  const float* c1_b = (const float*)d_in[21];
  const float* c2_w = (const float*)d_in[22];  const float* c2_b = (const float*)d_in[23];
  const float* r00_w = (const float*)d_in[24]; const float* r00_b = (const float*)d_in[25];
  const float* r01_w = (const float*)d_in[26]; const float* r01_b = (const float*)d_in[27];
  const float* r0c_w = (const float*)d_in[28]; const float* r0c_b = (const float*)d_in[29];
  const float* r10_w = (const float*)d_in[30]; const float* r10_b = (const float*)d_in[31];
  const float* r11_w = (const float*)d_in[32]; const float* r11_b = (const float*)d_in[33];
  const float* r1c_w = (const float*)d_in[34]; const float* r1c_b = (const float*)d_in[35];
  const float* c3_w = (const float*)d_in[36];  const float* c3_b = (const float*)d_in[37];
  const float* v1_w = (const float*)d_in[38];  const float* v1_b = (const float*)d_in[39];
  const float* v2_w = (const float*)d_in[40];  const float* v2_b = (const float*)d_in[41];
  const float* v3_w = (const float*)d_in[42];  const float* v3_b = (const float*)d_in[43];
  const float* out_w = (const float*)d_in[44]; const float* out_b = (const float*)d_in[45];

  // ---- workspace arena (f16, NP-row interleaved regions; byte-level aliasing)
  _Float16* F = (_Float16*)d_ws;
  _Float16* featbuf = F;                 // 4 x [NP][32]
  _Float16* fe1tmp  = F + 128 * PL;      // [NP][32]
  _Float16* RB      = F + 160 * PL;      // [NP][256] : [y2|rdb0|t0'|t1']
  _Float16* xc      = RB;                // [NP][128] aliases first half (dead later)
  _Float16* xcp     = RB + 128 * PL;     // [NP][128] aliases second half
  _Float16* Y1      = F + 416 * PL;      // [NP][64]; later aliased by ex [NP][48]
  _Float16* exb     = Y1;
  _Float16* RA      = F + 480 * PL;      // [NP][192] : [y2|t0|t1] -> [y2|rdb0|rdb1]
  float*    inp     = (float*)(F + 672 * PL);               // 4 f32 planes
  _Float16* WP      = (_Float16*)((char*)inp + 4 * PL * sizeof(float));

  _Float16* wp_fe2[4];
  _Float16* wcur = WP;
  for (int i = 0; i < 4; ++i) { wp_fe2[i] = wcur; wcur += 32 * 288; }
  _Float16* wp_pr  = wcur; wcur += 128 * 128;
  _Float16* wp_c1  = wcur; wcur += 64 * 1152;
  _Float16* wp_c2  = wcur; wcur += 64 * 576;
  _Float16* wp_r00 = wcur; wcur += 64 * 576;
  _Float16* wp_r01 = wcur; wcur += 64 * 1152;
  _Float16* wp_r0c = wcur; wcur += 64 * 192;
  _Float16* wp_r10 = wcur; wcur += 64 * 1152;
  _Float16* wp_r11 = wcur; wcur += 64 * 1728;
  _Float16* wp_r1c = wcur; wcur += 64 * 256;
  _Float16* wp_c3  = wcur; wcur += 64 * 192;
  _Float16* wp_v1  = wcur; wcur += 64 * 576;
  _Float16* wp_v2  = wcur; wcur += 64 * 64;
  _Float16* wp_v3  = wcur; wcur += 64 * 576;
  _Float16* wp_out = wcur; wcur += 48 * 64;

  auto packs = [&](const float* src, _Float16* dst, int Cout, int Cin, int KS, int Mpad) {
    int total = Mpad * KS * KS * Cin;
    pack_w_kernel<<<(total + 255) / 256, 256, 0, stream>>>(src, dst, Cout, Cin, KS, Mpad);
  };

  for (int i = 0; i < 4; ++i) packs((const float*)d_in[5 + 4 * i], wp_fe2[i], 32, 32, 3, 32);
  pack_pr_kernel<<<(128 * 128 + 255) / 256, 256, 0, stream>>>(pr_w, vptr, wp_pr);
  packs(c1_w, wp_c1, 64, 128, 3, 64);
  packs(c2_w, wp_c2, 64, 64, 3, 64);
  packs(r00_w, wp_r00, 64, 64, 3, 64);
  packs(r01_w, wp_r01, 64, 128, 3, 64);
  packs(r0c_w, wp_r0c, 64, 192, 1, 64);
  packs(r10_w, wp_r10, 64, 128, 3, 64);
  packs(r11_w, wp_r11, 64, 192, 3, 64);
  packs(r1c_w, wp_r1c, 64, 256, 1, 64);
  packs(c3_w, wp_c3, 64, 192, 1, 64);
  packs(v1_w, wp_v1, 64, 64, 3, 64);
  packs(v2_w, wp_v2, 64, 64, 1, 64);
  packs(v3_w, wp_v3, 64, 64, 3, 64);
  packs(out_w, wp_out, 36, 64, 1, 48);

  const dim3 g2(NP / 256), g1(NP / 128), blk(256);  // NTILES=2 / NTILES=1 grids
  const dim3 pgrid(NP / 256), pblk(256);

  // ---- per-frame feature extraction (conv1 VALU, conv2 WMMA)
  for (int i = 0; i < 4; ++i) {
    fe1_kernel<<<pgrid, pblk, 0, stream>>>(x_all + (size_t)i * PL,
                                           (const float*)d_in[3 + 4 * i],
                                           (const float*)d_in[4 + 4 * i], fe1tmp);
    conv_wmma_kernel<32, 2, 2, 3, true><<<g2, blk, SMEMB(2, 288), stream>>>(
        fe1tmp, 32, 0, wp_fe2[i], (const float*)d_in[6 + 4 * i],
        featbuf + (size_t)i * PL * 32, 32, 0, 32);
  }

  // ---- warp feats -> xc [NP][128] and x -> inp (4 f32 planes)
  warp_kernel<<<dim3(NP / 256, 4), 256, 0, stream>>>(x_all, flows, vptr, featbuf, xc, inp);

  // ---- trunk
  conv_wmma_kernel<128, 8, 1, 1, false><<<g1, blk, SMEMB(8, 128), stream>>>(
      xc, 128, 0, wp_pr, nullptr, xcp, 128, 0, 128);
  conv_wmma_kernel<128, 4, 2, 3, true><<<g2, blk, SMEMB(4, 1152), stream>>>(
      xcp, 128, 0, wp_c1, c1_b, Y1, 64, 0, 64);
  conv_wmma_kernel<64, 4, 2, 3, true><<<g2, blk, SMEMB(4, 576), stream>>>(
      Y1, 64, 0, wp_c2, c2_b, RA, 192, 0, 64);
  // y2 copy into RB cols 0-63 (xc bytes now dead)
  copy_cols_kernel<<<pgrid, pblk, 0, stream>>>(RA, 192, 0, RB, 256, 0, 64);
  // RDN0 on RA = [y2|t0|t1]
  conv_wmma_kernel<64, 4, 2, 3, true><<<g2, blk, SMEMB(4, 576), stream>>>(
      RA, 192, 0, wp_r00, r00_b, RA, 192, 64, 64);
  conv_wmma_kernel<128, 4, 2, 3, true><<<g2, blk, SMEMB(4, 1152), stream>>>(
      RA, 192, 0, wp_r01, r01_b, RA, 192, 128, 64);
  conv_wmma_kernel<192, 4, 2, 1, true><<<g2, blk, SMEMB(4, 192), stream>>>(
      RA, 192, 0, wp_r0c, r0c_b, RB, 256, 64, 64);
  // rdb0out copy into RA cols 64-127 (t0 dead) -> RA = [y2|rdb0out|t1]
  copy_cols_kernel<<<pgrid, pblk, 0, stream>>>(RB, 256, 64, RA, 192, 64, 64);
  // RDN1 on RB = [y2|rdb0out|t0'|t1']
  conv_wmma_kernel<128, 4, 2, 3, true><<<g2, blk, SMEMB(4, 1152), stream>>>(
      RB, 256, 0, wp_r10, r10_b, RB, 256, 128, 64);
  conv_wmma_kernel<192, 4, 2, 3, true><<<g2, blk, SMEMB(4, 1728), stream>>>(
      RB, 256, 0, wp_r11, r11_b, RB, 256, 192, 64);
  conv_wmma_kernel<256, 4, 2, 1, true><<<g2, blk, SMEMB(4, 256), stream>>>(
      RB, 256, 0, wp_r1c, r1c_b, RA, 192, 128, 64);
  // concat3 on RA = [y2|rdb0out|rdb1out]
  conv_wmma_kernel<192, 4, 2, 1, true><<<g2, blk, SMEMB(4, 192), stream>>>(
      RA, 192, 0, wp_c3, c3_b, RB, 256, 0, 64);
  conv_wmma_kernel<64, 4, 2, 3, true><<<g2, blk, SMEMB(4, 576), stream>>>(
      RB, 256, 0, wp_v1, v1_b, RB, 256, 64, 64);
  conv_wmma_kernel<64, 4, 2, 1, true><<<g2, blk, SMEMB(4, 64), stream>>>(
      RB, 256, 64, wp_v2, v2_b, RB, 256, 128, 64);
  conv_wmma_kernel<64, 4, 2, 3, true><<<g2, blk, SMEMB(4, 576), stream>>>(
      RB, 256, 128, wp_v3, v3_b, RB, 256, 192, 64);
  conv_wmma_kernel<64, 3, 2, 1, false><<<g2, blk, SMEMB(3, 64), stream>>>(
      RB, 256, 192, wp_out, out_b, exb, 48, 0, 36);

  // ---- dynamic filter -> output (B,1,H,W) f32
  dynfilter_kernel<<<pgrid, pblk, 0, stream>>>(exb, inp, (float*)d_out);
}
